// SelfAttention_54219667144861
// MI455X (gfx1250) — matrix operand order repaired and checked by reading, not compiled, not run on hardware
//
#include <hip/hip_runtime.h>
#include <hip/hip_bf16.h>
#include <stdint.h>

// Self-attention for MI455X (gfx1250, wave32, WMMA).
// S=2048 B=4 D=1024: ~190 GFLOP vs ~100MB traffic -> compute bound.
// All GEMM stages on v_wmma_f32_16x16x32_bf16. GEMM kernels use 32x64
// per-wave tiles (2 A strips x 4 N tiles) so each B-fragment load feeds two
// WMMAs, with K-chunked LDS staging (32KB) of the A strip shared by 8 waves.

#define S_LEN 2048
#define B_SZ  4
#define D_DIM 1024
#define KCH   512          // K-chunk staged in LDS: 32 rows x 512 = 32 KB

typedef __bf16 bf16_t;
typedef __attribute__((ext_vector_type(16))) __bf16 v16bf;
typedef __attribute__((ext_vector_type(8)))  __bf16 v8bf;
typedef __attribute__((ext_vector_type(2)))  __bf16 v2bf;
typedef __attribute__((ext_vector_type(8)))  float  v8f;

union BF16x16 { v16bf v; v8bf h[2]; };

// Packed fp32->bf16 (RNE), 8 at a time; prefers v_cvt_pk_bf16_f32.
__device__ __forceinline__ v8bf cvt8(const float* __restrict__ p) {
  v8bf o;
#if __has_builtin(__builtin_amdgcn_cvt_pk_bf16_f32)
#pragma unroll
  for (int i = 0; i < 4; ++i) {
    v2bf t = __builtin_amdgcn_cvt_pk_bf16_f32(p[2 * i], p[2 * i + 1]);
    o[2 * i] = t[0]; o[2 * i + 1] = t[1];
  }
#else
#pragma unroll
  for (int i = 0; i < 8; ++i) o[i] = (bf16_t)p[i];
#endif
  return o;
}

// ---- WMMA fragment loaders (layouts per CDNA5 ISA 7.12.2) ----
// A (16x32 bf16): lane L holds row M=L&15; halves at K=k0+8*hi, k0+16+8*hi.
__device__ __forceinline__ v16bf a_frag_bf16(const bf16_t* rowp, int hi, int k0) {
  BF16x16 f;
  f.h[0] = *(const v8bf*)(rowp + k0 + 8 * hi);
  f.h[1] = *(const v8bf*)(rowp + k0 + 16 + 8 * hi);
  return f.v;
}
// B (32x16 bf16): lane L holds col N=L&15; halves at K=k0+8*hi, k0+16+8*hi.
__device__ __forceinline__ v16bf b_frag_bf16(const bf16_t* rowp, int hi, int k0) {
  const bf16_t* p = rowp + k0 + 8 * hi;
  BF16x16 f;
  f.h[0] = *(const v8bf*)(p);
  f.h[1] = *(const v8bf*)(p + 16);
  return f.v;
}

// =================== Pack kernels (one-shot conversions) ====================
__global__ __launch_bounds__(256) void pack_x_kernel(
    const float* __restrict__ x, bf16_t* __restrict__ xb) {
  int g = blockIdx.x * 256 + threadIdx.x;
  const int rowlen = D_DIM / 8;
  int row = g / rowlen, c8 = g % rowlen;         // row over (s,b)
  int s = row / B_SZ, b = row % B_SZ;
  v8bf o = cvt8(x + (size_t)row * D_DIM + c8 * 8);
  *(v8bf*)(xb + ((size_t)b * S_LEN + s) * D_DIM + (size_t)c8 * 8) = o;
}

__global__ __launch_bounds__(256) void pack_w_kernel(
    const float* __restrict__ Wq, const float* __restrict__ Wk,
    const float* __restrict__ Wv, bf16_t* __restrict__ Wqb,
    bf16_t* __restrict__ Wkb, bf16_t* __restrict__ Wvb) {
  const float* src; bf16_t* dst;
  if (blockIdx.z == 0)      { src = Wq; dst = Wqb; }
  else if (blockIdx.z == 1) { src = Wk; dst = Wkb; }
  else                      { src = Wv; dst = Wvb; }
  size_t g = (size_t)(blockIdx.x * 256 + threadIdx.x) * 8;
  *(v8bf*)(dst + g) = cvt8(src + g);
}

// ================= Kernel 1: QKV projection GEMMs (grid.z = 0/1/2) ==========
// Per wave: 32x64 output tile (2 A strips x 4 N tiles, 8 accumulators).
// Block: 8 waves = 32 rows x 512 cols. A strip staged in LDS in 512-K chunks.
__global__ __launch_bounds__(256) void qkv_kernel(
    const bf16_t* __restrict__ xb,
    const bf16_t* __restrict__ Wqb, const float* __restrict__ bq,
    const bf16_t* __restrict__ Wkb, const float* __restrict__ bk,
    const bf16_t* __restrict__ Wvb, const float* __restrict__ bv,
    bf16_t* __restrict__ q, bf16_t* __restrict__ k, bf16_t* __restrict__ vT) {
  const int lane = threadIdx.x & 31;
  const int wave = threadIdx.x >> 5;              // 8 waves
  const int hi   = lane >> 4;
  const int r    = lane & 15;
  const int row0 = blockIdx.x * 32;               // over flattened (B,S)
  const int col0 = blockIdx.y * 512 + wave * 64;
  const int b    = row0 / S_LEN;
  const int s0   = row0 % S_LEN;

  const bf16_t* W; const float* bias;
  if (blockIdx.z == 0)      { W = Wqb; bias = bq; }
  else if (blockIdx.z == 1) { W = Wkb; bias = bk; }
  else                      { W = Wvb; bias = bv; }

  __shared__ bf16_t xtile[32 * KCH];              // 32 KB

  const bf16_t* brow[4];
  v8f acc[2][4];
#pragma unroll
  for (int t = 0; t < 4; ++t) {
    brow[t] = W + (size_t)(col0 + t * 16 + r) * D_DIM;  // B[kd][n] = W[n][kd]
    float bb = bias[col0 + t * 16 + r];
#pragma unroll
    for (int m = 0; m < 2; ++m)
#pragma unroll
      for (int i = 0; i < 8; ++i) acc[m][t][i] = bb;
  }

  for (int kc = 0; kc < D_DIM; kc += KCH) {
    __syncthreads();                               // retire reads of prev chunk
    for (int i = threadIdx.x; i < 32 * KCH / 8; i += 256) {
      int row = i >> 6, c8 = i & 63;               // 64 v8bf per row
      ((v8bf*)xtile)[i] =
          *(const v8bf*)(xb + (size_t)(row0 + row) * D_DIM + kc + c8 * 8);
    }
    __syncthreads();
    const bf16_t* arow0 = xtile + (size_t)r * KCH;
    const bf16_t* arow1 = xtile + (size_t)(16 + r) * KCH;
    for (int k0 = 0; k0 < KCH; k0 += 32) {
      __builtin_prefetch(brow[0] + kc + k0 + 64, 0, 3);
      v16bf a0 = a_frag_bf16(arow0, hi, k0);
      v16bf a1 = a_frag_bf16(arow1, hi, k0);
#pragma unroll
      for (int t = 0; t < 4; ++t) {
        v16bf bf = b_frag_bf16(brow[t], hi, kc + k0);
        acc[0][t] = __builtin_amdgcn_wmma_f32_16x16x32_bf16(
            false, a0, false, bf, (short)0, acc[0][t], false, false);
        acc[1][t] = __builtin_amdgcn_wmma_f32_16x16x32_bf16(
            false, a1, false, bf, (short)0, acc[1][t], false, false);
      }
    }
  }

  if (blockIdx.z < 2) {
    bf16_t* out = (blockIdx.z == 0) ? q : k;
#pragma unroll
    for (int m = 0; m < 2; ++m)
#pragma unroll
      for (int t = 0; t < 4; ++t) {
        size_t base = (size_t)(row0 + 16 * m + 8 * hi) * D_DIM
                      + col0 + t * 16 + r;
#pragma unroll
        for (int i = 0; i < 8; ++i)
          out[base + (size_t)i * D_DIM] = (bf16_t)acc[m][t][i];
      }
  } else {
#pragma unroll
    for (int m = 0; m < 2; ++m)
#pragma unroll
      for (int t = 0; t < 4; ++t) {        // transposed store: contiguous in s
        v8bf pk;
#pragma unroll
        for (int i = 0; i < 8; ++i) pk[i] = (bf16_t)acc[m][t][i];
        bf16_t* dst = vT + ((size_t)b * D_DIM + col0 + t * 16 + r) * S_LEN
                         + s0 + 16 * m + 8 * hi;
        *(v8bf*)dst = pk;
      }
  }
}

// ====== Kernel 2: scores + softmax numerator -> P (unnormalized), row sums ===
__global__ __launch_bounds__(256) void attn_softmax_kernel(
    const bf16_t* __restrict__ q, const bf16_t* __restrict__ kmat,
    bf16_t* __restrict__ P, float* __restrict__ lsum) {
  const int lane = threadIdx.x & 31;
  const int wave = threadIdx.x >> 5;   // 8 waves
  const int hi   = lane >> 4;
  const int r    = lane & 15;
  const int q0   = blockIdx.x * 16;
  const int b    = blockIdx.y;

  __shared__ bf16_t qtile[16 * D_DIM];   // 32 KB query strip, reused 256x
  __shared__ float red[8][16];
  __shared__ float gmax[16];

  const bf16_t* kb = kmat + (size_t)b * S_LEN * D_DIM;
  bf16_t* Pb = P + (size_t)b * S_LEN * S_LEN;
  const float scale = 0.03125f;        // 1/sqrt(1024)

  {
    const v8bf* src = (const v8bf*)(q + ((size_t)b * S_LEN + q0) * D_DIM);
    v8bf* dst = (v8bf*)qtile;
    for (int i = threadIdx.x; i < 16 * D_DIM / 8; i += 256) dst[i] = src[i];
  }
  __syncthreads();
  const bf16_t* arow = qtile + (size_t)r * D_DIM;

  // ---- sweep 1: row max ----
  float rmax[8];
#pragma unroll
  for (int i = 0; i < 8; ++i) rmax[i] = -3.0e38f;

  for (int kt = wave; kt < S_LEN / 16; kt += 8) {
    const bf16_t* brow = kb + (size_t)(kt * 16 + r) * D_DIM; // B[kd][n]=K[n][kd]
    v8f c = {};
    for (int k0 = 0; k0 < D_DIM; k0 += 32) {
      v16bf a = a_frag_bf16(arow, hi, k0);
      v16bf bf = b_frag_bf16(brow, hi, k0);
      c = __builtin_amdgcn_wmma_f32_16x16x32_bf16(
          false, a, false, bf, (short)0, c, false, false);
    }
#pragma unroll
    for (int i = 0; i < 8; ++i) rmax[i] = fmaxf(rmax[i], c[i] * scale);
  }
#pragma unroll
  for (int m = 1; m < 16; m <<= 1)
#pragma unroll
    for (int i = 0; i < 8; ++i)
      rmax[i] = fmaxf(rmax[i], __shfl_xor(rmax[i], m, 32));
  if (r == 0) {
#pragma unroll
    for (int i = 0; i < 8; ++i) red[wave][hi * 8 + i] = rmax[i];
  }
  __syncthreads();
  if (threadIdx.x < 16) {
    float m = red[0][threadIdx.x];
#pragma unroll
    for (int w = 1; w < 8; ++w) m = fmaxf(m, red[w][threadIdx.x]);
    gmax[threadIdx.x] = m;
  }
  __syncthreads();

  // ---- sweep 2: P = exp(s - max), accumulate row sums ----
  float rowm[8], rsum[8];
#pragma unroll
  for (int i = 0; i < 8; ++i) { rowm[i] = gmax[hi * 8 + i]; rsum[i] = 0.0f; }

  for (int kt = wave; kt < S_LEN / 16; kt += 8) {
    const bf16_t* brow = kb + (size_t)(kt * 16 + r) * D_DIM;
    v8f c = {};
    for (int k0 = 0; k0 < D_DIM; k0 += 32) {
      v16bf a = a_frag_bf16(arow, hi, k0);
      v16bf bf = b_frag_bf16(brow, hi, k0);
      c = __builtin_amdgcn_wmma_f32_16x16x32_bf16(
          false, a, false, bf, (short)0, c, false, false);
    }
    size_t base = (size_t)(q0 + 8 * hi) * S_LEN + kt * 16 + r;
#pragma unroll
    for (int i = 0; i < 8; ++i) {
      float e = __expf(c[i] * scale - rowm[i]);
      rsum[i] += e;
      Pb[base + (size_t)i * S_LEN] = (bf16_t)e;
    }
  }
#pragma unroll
  for (int m = 1; m < 16; m <<= 1)
#pragma unroll
    for (int i = 0; i < 8; ++i) rsum[i] += __shfl_xor(rsum[i], m, 32);
  if (r == 0) {
#pragma unroll
    for (int i = 0; i < 8; ++i) red[wave][hi * 8 + i] = rsum[i];
  }
  __syncthreads();
  if (threadIdx.x < 16) {
    float s = 0.0f;
#pragma unroll
    for (int w = 0; w < 8; ++w) s += red[w][threadIdx.x];
    lsum[(size_t)b * S_LEN + q0 + threadIdx.x] = s;
  }
}

// ========== Kernel 3: O = (P @ V) / rowsum, fp32 out in (B,S,D) flat ========
// Per wave 32x64 tile; P strip staged in LDS in 512-K chunks (32 KB).
__global__ __launch_bounds__(256) void out_gemm_kernel(
    const bf16_t* __restrict__ P, const bf16_t* __restrict__ vT,
    const float* __restrict__ lsum, float* __restrict__ out) {
  const int lane = threadIdx.x & 31;
  const int wave = threadIdx.x >> 5;              // 8 waves
  const int hi   = lane >> 4;
  const int r    = lane & 15;
  const int row0 = blockIdx.x * 32;               // over flattened (B,S)
  const int col0 = blockIdx.y * 512 + wave * 64;
  const int b    = row0 / S_LEN;
  const int s0   = row0 % S_LEN;

  __shared__ bf16_t ptile[32 * KCH];              // 32 KB
  const bf16_t* Pb = P  + (size_t)b * S_LEN * S_LEN;
  const bf16_t* vb = vT + (size_t)b * D_DIM * S_LEN;

  const bf16_t* brow[4];
  v8f acc[2][4];
#pragma unroll
  for (int t = 0; t < 4; ++t) {
    brow[t] = vb + (size_t)(col0 + t * 16 + r) * S_LEN; // B[kk][n] = vT[n][kk]
#pragma unroll
    for (int m = 0; m < 2; ++m) acc[m][t] = (v8f){};
  }

  for (int kc = 0; kc < S_LEN; kc += KCH) {
    __syncthreads();
    for (int i = threadIdx.x; i < 32 * KCH / 8; i += 256) {
      int row = i >> 6, c8 = i & 63;
      ((v8bf*)ptile)[i] =
          *(const v8bf*)(Pb + (size_t)(s0 + row) * S_LEN + kc + c8 * 8);
    }
    __syncthreads();
    const bf16_t* arow0 = ptile + (size_t)r * KCH;
    const bf16_t* arow1 = ptile + (size_t)(16 + r) * KCH;
    for (int k0 = 0; k0 < KCH; k0 += 32) {
      __builtin_prefetch(brow[0] + kc + k0 + 64, 0, 3);
      v16bf a0 = a_frag_bf16(arow0, hi, k0);
      v16bf a1 = a_frag_bf16(arow1, hi, k0);
#pragma unroll
      for (int t = 0; t < 4; ++t) {
        v16bf bf = b_frag_bf16(brow[t], hi, kc + k0);
        acc[0][t] = __builtin_amdgcn_wmma_f32_16x16x32_bf16(
            false, a0, false, bf, (short)0, acc[0][t], false, false);
        acc[1][t] = __builtin_amdgcn_wmma_f32_16x16x32_bf16(
            false, a1, false, bf, (short)0, acc[1][t], false, false);
      }
    }
  }

  float inv[2][8];
#pragma unroll
  for (int m = 0; m < 2; ++m)
#pragma unroll
    for (int i = 0; i < 8; ++i)
      inv[m][i] = 1.0f / lsum[(size_t)b * S_LEN + s0 + 16 * m + 8 * hi + i];

#pragma unroll
  for (int m = 0; m < 2; ++m)
#pragma unroll
    for (int t = 0; t < 4; ++t)
#pragma unroll
      for (int i = 0; i < 8; ++i) {
        size_t idx = ((size_t)b * S_LEN + s0 + 16 * m + 8 * hi + i) * D_DIM
                     + col0 + t * 16 + r;
        out[idx] = acc[m][t][i] * inv[m][i];
      }
}

extern "C" void kernel_launch(void* const* d_in, const int* in_sizes, int n_in,
                              void* d_out, int out_size, void* d_ws, size_t ws_size,
                              hipStream_t stream) {
  const float* x  = (const float*)d_in[0];
  const float* Wq = (const float*)d_in[1];
  const float* bq = (const float*)d_in[2];
  const float* Wk = (const float*)d_in[3];
  const float* bk = (const float*)d_in[4];
  const float* Wv = (const float*)d_in[5];
  const float* bv = (const float*)d_in[6];
  float* out = (float*)d_out;

  // Workspace (bytes): xb 16MB | Wqb/Wkb/Wvb 2MB ea | q 16 | k 16 | vT 16 |
  //                    P 32MB | lsum 32KB   (total ~102MB)
  char* ws = (char*)d_ws;
  const size_t MB = 1024u * 1024u;
  bf16_t* xb  = (bf16_t*)(ws);
  bf16_t* Wqb = (bf16_t*)(ws + 16 * MB);
  bf16_t* Wkb = (bf16_t*)(ws + 18 * MB);
  bf16_t* Wvb = (bf16_t*)(ws + 20 * MB);
  bf16_t* q   = (bf16_t*)(ws + 22 * MB);
  bf16_t* k   = (bf16_t*)(ws + 38 * MB);
  bf16_t* vT  = (bf16_t*)(ws + 54 * MB);
  bf16_t* P   = (bf16_t*)(ws + 70 * MB);
  float*  l   = (float*) (ws + 102 * MB);

  pack_x_kernel<<<dim3((S_LEN * B_SZ * D_DIM / 8) / 256), 256, 0, stream>>>(x, xb);
  pack_w_kernel<<<dim3((D_DIM * D_DIM / 8) / 256, 1, 3), 256, 0, stream>>>(
      Wq, Wk, Wv, Wqb, Wkb, Wvb);
  // 1) QKV: 256 row tiles (32 rows) x 2 col groups (512) x 3 matrices
  qkv_kernel<<<dim3(256, 2, 3), 256, 0, stream>>>(xb, Wqb, bq, Wkb, bk, Wvb, bv,
                                                  q, k, vT);
  // 2) scores + softmax numerator: 128 query tiles x B
  attn_softmax_kernel<<<dim3(128, 4), 256, 0, stream>>>(q, k, P, l);
  // 3) O = P@V / rowsum: 256 row tiles x 2 col groups
  out_gemm_kernel<<<dim3(256, 2), 256, 0, stream>>>(P, vT, l, out);
}
